// MultiHeadAttention_10453950399076
// MI455X (gfx1250) — matrix operand order repaired and checked by reading, not compiled, not run on hardware
//
#include <hip/hip_runtime.h>

// ---------------------------------------------------------------------------
// Types
// ---------------------------------------------------------------------------
typedef __bf16 bf16_t;
typedef __attribute__((ext_vector_type(16))) __bf16   v16bf;
typedef __attribute__((ext_vector_type(2)))  __bf16   v2bf;
typedef __attribute__((ext_vector_type(8)))  float    v8f;
typedef __attribute__((ext_vector_type(4)))  unsigned u32x4;
typedef __attribute__((ext_vector_type(4)))  float    f32x4;

union V16 {
  v16bf  v;
  u32x4  q[2];
  v2bf   p[8];
  bf16_t e[16];
};

union V8pack {
  u32x4  q;
  v2bf   p[4];
  bf16_t e[8];
};

__device__ __forceinline__ bf16_t f2bf(float f) {
  unsigned u = __builtin_bit_cast(unsigned, f);
  unsigned r = u + 0x7FFFu + ((u >> 16) & 1u);           // round-to-nearest-even
  unsigned short s = (unsigned short)(r >> 16);
  return __builtin_bit_cast(bf16_t, s);
}

__device__ __forceinline__ v2bf f2bf2(float lo, float hi) {
#if __has_builtin(__builtin_amdgcn_cvt_pk_bf16_f32)
  return __builtin_amdgcn_cvt_pk_bf16_f32(lo, hi);
#else
  v2bf r;
  r[0] = f2bf(lo);
  r[1] = f2bf(hi);
  return r;
#endif
}

__device__ __forceinline__ v8f wmma_bf16(v16bf a, v16bf b, v8f c) {
  return __builtin_amdgcn_wmma_f32_16x16x32_bf16(false, a, false, b, (short)0, c,
                                                 false, false);
}

// Problem sizes: b=4, n=2048, d=1024, M=8192, scale = 1/sqrt(1024) = 0.03125

// ---------------------------------------------------------------------------
// Weight prep: W [out=1024][in=1024] fp32 -> W^T bf16 in B-operand swizzle.
// B-tile (32x16): lane = ((k&31)>>4)*16 + (n&15), slot = k&15, 512/block.
// ---------------------------------------------------------------------------
__global__ void prep_weights(const float* __restrict__ Wq, const float* __restrict__ Wk,
                             const float* __restrict__ Wv, const float* __restrict__ Wo,
                             bf16_t* __restrict__ dst) {
  int idx = blockIdx.x * 256 + threadIdx.x;   // 4 * 1M elements
  int w   = idx >> 20;
  int rem = idx & 0xFFFFF;
  int n   = rem >> 10;        // output feature
  int k   = rem & 1023;       // input feature (consecutive -> coalesced read)
  const float* W = (w == 0) ? Wq : (w == 1) ? Wk : (w == 2) ? Wv : Wo;
  float val = W[n * 1024 + k];                // B[k][n] = (W^T)[k][n]
  int lane = (((k >> 4) & 1) << 4) | (n & 15);
  int slot = k & 15;
  int blk  = (k >> 5) * 64 + (n >> 4);
  dst[((size_t)w << 20) + (size_t)blk * 512 + lane * 16 + slot] = f2bf(val);
}

// ---------------------------------------------------------------------------
// Projection GEMM: out = X @ W^T + bias  (X fp32 [8192,1024], W^T swizzled bf16)
// Block: 16 rows; wave w owns cols [w*128, w*128+128).
// MODE 0: qp -> A-swizzle / batch.  MODE 1: kp -> B-swizzle (K=d,N=n).
// MODE 2: vp -> B-swizzle (K=n,N=d).
// ---------------------------------------------------------------------------
template <int MODE>
__global__ void __launch_bounds__(256) proj_kernel(const float* __restrict__ X,
                                                   const bf16_t* __restrict__ Bw,
                                                   const float* __restrict__ bias,
                                                   bf16_t* __restrict__ out) {
  const int m0     = blockIdx.x << 4;
  const int lane   = threadIdx.x & 31;
  const int wave   = threadIdx.x >> 5;
  const int lane15 = lane & 15;
  const int hi     = lane >> 4;
  const int n0w    = wave << 7;

  const float* xrow = X + (size_t)(m0 + lane15) * 1024;

  v8f acc[8];
  for (int t = 0; t < 8; ++t)
    for (int r = 0; r < 8; ++r) acc[t][r] = 0.0f;

  for (int k0 = 0; k0 < 1024; k0 += 32) {
    // prefetch next k-panel (A row segment + the 8 consecutive B blocks)
    if (k0 + 32 < 1024) {
      __builtin_prefetch((const void*)(xrow + k0 + 32), 0, 1);
      __builtin_prefetch(
          (const void*)(Bw + (size_t)(((k0 + 32) >> 5) * 64 + (n0w >> 4)) * 512 +
                        lane * 128),
          0, 1);
    }
    // A tile 16x32
    V16 a;
    f32x4 f0 = *(const f32x4*)(xrow + k0 + hi * 16);
    f32x4 f1 = *(const f32x4*)(xrow + k0 + hi * 16 + 4);
    f32x4 f2 = *(const f32x4*)(xrow + k0 + hi * 16 + 8);
    f32x4 f3 = *(const f32x4*)(xrow + k0 + hi * 16 + 12);
    a.p[0] = f2bf2(f0[0], f0[1]);
    a.p[1] = f2bf2(f0[2], f0[3]);
    a.p[2] = f2bf2(f1[0], f1[1]);
    a.p[3] = f2bf2(f1[2], f1[3]);
    a.p[4] = f2bf2(f2[0], f2[1]);
    a.p[5] = f2bf2(f2[2], f2[3]);
    a.p[6] = f2bf2(f3[0], f3[1]);
    a.p[7] = f2bf2(f3[2], f3[3]);
    // batch-load all 8 B tiles (one s_clause), then run the WMMA chain
    V16 b[8];
#pragma unroll
    for (int t = 0; t < 8; ++t) {
      const int c0 = n0w + (t << 4);
      const u32x4* bp =
          (const u32x4*)(Bw + (size_t)((k0 >> 5) * 64 + (c0 >> 4)) * 512) + lane * 2;
      b[t].q[0] = bp[0];
      b[t].q[1] = bp[1];
    }
#pragma unroll
    for (int t = 0; t < 8; ++t) acc[t] = wmma_bf16(a.v, b[t].v, acc[t]);
  }

  const int batch = m0 >> 11;
  const int m0l   = m0 & 2047;
  bf16_t* ob = out + (size_t)batch * (2048 * 1024);

#pragma unroll
  for (int t = 0; t < 8; ++t) {
    const int c  = n0w + (t << 4) + lane15;
    const float bv = bias[c];
    if (MODE == 0) {
      // qp, A-swizzle: element (m, k=c)
      const int s   = c & 15;
      const int lp  = (((c >> 4) & 1) << 4) + 8 * hi;
      const int blk = (m0l >> 4) * 32 + (c >> 5);
      bf16_t* p = ob + (size_t)blk * 512 + lp * 16 + s;
#pragma unroll
      for (int r = 0; r < 8; ++r) p[r * 16] = f2bf(acc[t][r] + bv);
    } else if (MODE == 1) {
      // kp, B-swizzle (K=d, N=n): element (k=c, n=m)
      const int s   = c & 15;
      const int lp  = (((c >> 4) & 1) << 4) + 8 * hi;
      const int blk = (c >> 5) * 128 + (m0l >> 4);
      bf16_t* p = ob + (size_t)blk * 512 + lp * 16 + s;
#pragma unroll
      for (int r = 0; r < 8; ++r) p[r * 16] = f2bf(acc[t][r] + bv);
    } else {
      // vp, B-swizzle (K=n, N=d): element (k=m, n=c) -> one b128 store
      const int krb = (m0l & 16) + 8 * hi;
      const int lp  = (((krb >> 4) & 1) << 4) + (c & 15);
      const int sb  = krb & 8;
      const int blk = (m0l >> 5) * 64 + (c >> 4);
      V8pack pk;
#pragma unroll
      for (int r = 0; r < 4; ++r)
        pk.p[r] = f2bf2(acc[t][2 * r] + bv, acc[t][2 * r + 1] + bv);
      *(u32x4*)(ob + (size_t)blk * 512 + lp * 16 + sb) = pk.q;
    }
  }
}

// ---------------------------------------------------------------------------
// Flash attention, BK=64 key blocks. Workgroup = 8 waves owns 16 query rows.
// Branch-free inner loop: all 4 key subtiles are always computed; causal
// masking is applied only in the softmax (masked P entries are exact zeros,
// so the extra WMMAs on the final block are harmless).
// Output is d-major (yT) so the reference's transpose bug is free.
// ---------------------------------------------------------------------------
__global__ void __launch_bounds__(256) attn_kernel(const bf16_t* __restrict__ qp,
                                                   const bf16_t* __restrict__ kpw,
                                                   const bf16_t* __restrict__ vpw,
                                                   bf16_t* __restrict__ yT) {
  const int batch  = blockIdx.x >> 7;
  const int q0     = (blockIdx.x & 127) << 4;
  const int tid    = threadIdx.x;
  const int lane   = tid & 31;
  const int lane15 = lane & 15;
  const int hi     = lane >> 4;
  const int n0w    = (tid >> 5) << 7;

  __shared__ float sS[1024];      // 16 rows x 64 keys fp32
  __shared__ u32x4 sPq[128];      // 1024 bf16: 2 A-swizzled P tiles
  __shared__ float sAlpha[16];
  __shared__ float sNorm[16];
  bf16_t* sP = (bf16_t*)sPq;

  const bf16_t* qb = qp  + (size_t)batch * (2048 * 1024);
  const bf16_t* kb = kpw + (size_t)batch * (1024 * 2048);
  const bf16_t* vb = vpw + (size_t)batch * (2048 * 1024);

  v8f acc[8];
  for (int t = 0; t < 8; ++t)
    for (int r = 0; r < 8; ++r) acc[t][r] = 0.0f;

  // Q tiles for this wave's d-slice are key-loop invariant: load once.
  V16 aq[4];
#pragma unroll
  for (int kk = 0; kk < 4; ++kk) {
    const int d0 = n0w + (kk << 5);
    const u32x4* ap =
        (const u32x4*)(qb + (size_t)((q0 >> 4) * 32 + (d0 >> 5)) * 512) + lane * 2;
    aq[kk].q[0] = ap[0];
    aq[kk].q[1] = ap[1];
  }

  float mrow = -3.0e38f;   // meaningful in lanes tid<16 only
  float lrow = 0.0f;

  const int nkb = (q0 + 16 + 63) >> 6;     // 64-key blocks

  for (int kbi = 0; kbi < nkb; ++kbi) {
    const int key0 = kbi << 6;

    sS[tid]       = 0.0f;
    sS[tid + 256] = 0.0f;
    sS[tid + 512] = 0.0f;
    sS[tid + 768] = 0.0f;
    __syncthreads();

    // ---- partial scores over this wave's 128-wide d-slice (branch-free) ----
    v8f ps[4];
#pragma unroll
    for (int s = 0; s < 4; ++s)
      for (int r = 0; r < 8; ++r) ps[s][r] = 0.0f;

#pragma unroll
    for (int kk = 0; kk < 4; ++kk) {
      const int d0 = n0w + (kk << 5);
      const u32x4* kbase =
          (const u32x4*)(kb + (size_t)((d0 >> 5) * 128 + (key0 >> 4)) * 512) + lane * 2;
      if (kbi + 1 < nkb)   // prefetch next key block of K^T for this d-chunk
        __builtin_prefetch((const void*)(kbase + 256), 0, 1);
      V16 b0, b1, b2, b3;
      b0.q[0] = kbase[0];   b0.q[1] = kbase[1];
      b1.q[0] = kbase[64];  b1.q[1] = kbase[65];
      b2.q[0] = kbase[128]; b2.q[1] = kbase[129];
      b3.q[0] = kbase[192]; b3.q[1] = kbase[193];
      ps[0] = wmma_bf16(aq[kk].v, b0.v, ps[0]);
      ps[1] = wmma_bf16(aq[kk].v, b1.v, ps[1]);
      ps[2] = wmma_bf16(aq[kk].v, b2.v, ps[2]);
      ps[3] = wmma_bf16(aq[kk].v, b3.v, ps[3]);
    }
#pragma unroll
    for (int r = 0; r < 8; ++r) {
      const int row = (8 * hi + r) * 64 + lane15;
      atomicAdd(&sS[row],      ps[0][r]);
      atomicAdd(&sS[row + 16], ps[1][r]);
      atomicAdd(&sS[row + 32], ps[2][r]);
      atomicAdd(&sS[row + 48], ps[3][r]);
    }
    __syncthreads();

    // ---- online softmax: lanes 0..15 of wave 0, one row each, 2 passes ----
    if (tid < 16) {
      const int r = tid;
      const int rlim = q0 + r;
      const f32x4* srow = (const f32x4*)(sS + r * 64);
      float mx = mrow;
#pragma unroll 4
      for (int jj = 0; jj < 16; ++jj) {
        const f32x4 s4 = srow[jj];
#pragma unroll
        for (int e = 0; e < 4; ++e) {
          const int j = jj * 4 + e;
          const float sv = (key0 + j <= rlim) ? s4[e] * 0.03125f : -1.0e30f;
          mx = fmaxf(mx, sv);
        }
      }
      float sum = 0.0f;
#pragma unroll 4
      for (int jj = 0; jj < 16; ++jj) {
        const f32x4 s4 = srow[jj];
        float p[4];
#pragma unroll
        for (int e = 0; e < 4; ++e) {
          const int j = jj * 4 + e;
          const float sv = (key0 + j <= rlim) ? s4[e] * 0.03125f : -1.0e30f;
          p[e] = __expf(sv - mx);
          sum += p[e];
        }
        // A-layout: slots for an aligned 4-run of j are contiguous
        const int j0  = jj * 4;
        const int j32 = j0 & 31;
        const int idx = ((j0 >> 5) << 9) + ((((j32 >> 4) & 1) << 4) + r) * 16 +
                        (j32 & 15);
        *(v2bf*)(sP + idx)     = f2bf2(p[0], p[1]);
        *(v2bf*)(sP + idx + 2) = f2bf2(p[2], p[3]);
      }
      const float alpha = __expf(mrow - mx);
      lrow = lrow * alpha + sum;
      mrow = mx;
      sAlpha[r] = alpha;
    }
    __syncthreads();

    // ---- rescale accumulators, then P(16x64) x V(64x128-slice) ----
    float al[8];
#pragma unroll
    for (int r = 0; r < 8; ++r) al[r] = sAlpha[8 * hi + r];
    V16 pa0, pa1;
    pa0.q[0] = ((const u32x4*)sP)[lane * 2];
    pa0.q[1] = ((const u32x4*)sP)[lane * 2 + 1];
    pa1.q[0] = ((const u32x4*)sP)[64 + lane * 2];
    pa1.q[1] = ((const u32x4*)sP)[64 + lane * 2 + 1];
#pragma unroll
    for (int t = 0; t < 8; ++t) {
      const int c0 = n0w + (t << 4);
      const u32x4* vbase =
          (const u32x4*)(vb + (size_t)((key0 >> 5) * 64 + (c0 >> 4)) * 512) + lane * 2;
      if (kbi + 1 < nkb)
        __builtin_prefetch((const void*)(vbase + 256), 0, 1);
      V16 b0, b1;
      b0.q[0] = vbase[0];
      b0.q[1] = vbase[1];
      b1.q[0] = vbase[128];
      b1.q[1] = vbase[129];
#pragma unroll
      for (int r = 0; r < 8; ++r) acc[t][r] *= al[r];
      acc[t] = wmma_bf16(pa0.v, b0.v, acc[t]);
      acc[t] = wmma_bf16(pa1.v, b1.v, acc[t]);
    }
    __syncthreads();
  }

  if (tid < 16) sNorm[tid] = (lrow > 0.0f) ? (1.0f / lrow) : 0.0f;
  __syncthreads();

  float nv[8];
  for (int r = 0; r < 8; ++r) nv[r] = sNorm[8 * hi + r];

  bf16_t* yb = yT + (size_t)batch * (1024 * 2048);
#pragma unroll
  for (int t = 0; t < 8; ++t) {
    const int c = n0w + (t << 4) + lane15;
    V8pack pk;
#pragma unroll
    for (int r = 0; r < 4; ++r)
      pk.p[r] = f2bf2(acc[t][2 * r] * nv[2 * r], acc[t][2 * r + 1] * nv[2 * r + 1]);
    *(u32x4*)(yb + (size_t)c * 2048 + q0 + 8 * hi) = pk.q;   // yT[b][c][m], b128
  }
}

// ---------------------------------------------------------------------------
// Output projection: out = y_bugged @ Wo^T + bo.  y_bugged == yT viewed flat
// [8192,1024] row-major (bug reproduced by construction).  fp32 output.
// ---------------------------------------------------------------------------
__global__ void __launch_bounds__(256) oproj_kernel(const bf16_t* __restrict__ A,
                                                    const bf16_t* __restrict__ Bw,
                                                    const float* __restrict__ bias,
                                                    float* __restrict__ out) {
  const int m0     = blockIdx.x << 4;
  const int lane   = threadIdx.x & 31;
  const int wave   = threadIdx.x >> 5;
  const int lane15 = lane & 15;
  const int hi     = lane >> 4;
  const int n0w    = wave << 7;

  const bf16_t* arow = A + (size_t)(m0 + lane15) * 1024;

  v8f acc[8];
  for (int t = 0; t < 8; ++t)
    for (int r = 0; r < 8; ++r) acc[t][r] = 0.0f;

  for (int k0 = 0; k0 < 1024; k0 += 32) {
    if (k0 + 32 < 1024) {
      __builtin_prefetch((const void*)(arow + k0 + 32), 0, 1);
      __builtin_prefetch(
          (const void*)(Bw + (size_t)(((k0 + 32) >> 5) * 64 + (n0w >> 4)) * 512 +
                        lane * 128),
          0, 1);
    }
    V16 a;
    a.q[0] = *(const u32x4*)(arow + k0 + hi * 16);
    a.q[1] = *(const u32x4*)(arow + k0 + hi * 16 + 8);
    V16 b[8];
#pragma unroll
    for (int t = 0; t < 8; ++t) {
      const int c0 = n0w + (t << 4);
      const u32x4* bp =
          (const u32x4*)(Bw + (size_t)((k0 >> 5) * 64 + (c0 >> 4)) * 512) + lane * 2;
      b[t].q[0] = bp[0];
      b[t].q[1] = bp[1];
    }
#pragma unroll
    for (int t = 0; t < 8; ++t) acc[t] = wmma_bf16(a.v, b[t].v, acc[t]);
  }

#pragma unroll
  for (int t = 0; t < 8; ++t) {
    const int c  = n0w + (t << 4) + lane15;
    const float bv = bias[c];
#pragma unroll
    for (int r = 0; r < 8; ++r)
      out[(size_t)(m0 + 8 * hi + r) * 1024 + c] = acc[t][r] + bv;
  }
}

// ---------------------------------------------------------------------------
// Launcher.  Workspace (bf16 elements):
//   0      : 4 x 1M swizzled W^T (q,k,v,o)     ( 8 MB)
//   +4M    : qp  (A-swizzled per batch)        (16 MB)
//   +12M   : kp  (B-swizzled, K=d N=n)         (16 MB)
//   +20M   : vp  (B-swizzled, K=n N=d)         (16 MB)
//   +28M   : yT  (d-major y == bugged view)    (16 MB)   total 72 MB
// ---------------------------------------------------------------------------
extern "C" void kernel_launch(void* const* d_in, const int* in_sizes, int n_in,
                              void* d_out, int out_size, void* d_ws, size_t ws_size,
                              hipStream_t stream) {
  (void)in_sizes; (void)n_in; (void)out_size; (void)ws_size;
  const float* q  = (const float*)d_in[0];
  const float* k  = (const float*)d_in[1];
  const float* v  = (const float*)d_in[2];
  const float* Wq = (const float*)d_in[3];
  const float* bq = (const float*)d_in[4];
  const float* Wk = (const float*)d_in[5];
  const float* bk = (const float*)d_in[6];
  const float* Wv = (const float*)d_in[7];
  const float* bv = (const float*)d_in[8];
  const float* Wo = (const float*)d_in[9];
  const float* bo = (const float*)d_in[10];

  bf16_t* ws = (bf16_t*)d_ws;
  bf16_t* wW = ws;                           // 4 x 1M elems
  bf16_t* qp = ws + ((size_t)4 << 20);
  bf16_t* kp = qp + ((size_t)8 << 20);
  bf16_t* vp = kp + ((size_t)8 << 20);
  bf16_t* yT = vp + ((size_t)8 << 20);

  prep_weights<<<(4 << 20) / 256, 256, 0, stream>>>(Wq, Wk, Wv, Wo, wW);
  proj_kernel<0><<<512, 256, 0, stream>>>(q, wW,                      bq, qp);
  proj_kernel<1><<<512, 256, 0, stream>>>(k, wW + ((size_t)1 << 20),  bk, kp);
  proj_kernel<2><<<512, 256, 0, stream>>>(v, wW + ((size_t)2 << 20),  bv, vp);
  attn_kernel<<<512, 256, 0, stream>>>(qp, kp, vp, yT);
  oproj_kernel<<<512, 256, 0, stream>>>(yT, wW + ((size_t)3 << 20), bo, (float*)d_out);
}
